// InteractionNetwork_57466662420972
// MI455X (gfx1250) — matrix operand order repaired and checked by reading, not compiled, not run on hardware
//
#include <hip/hip_runtime.h>
#include <hip/hip_bf16.h>
#include <hip/hip_fp16.h>

// ---------------------------------------------------------------------------
// InteractionNetwork on MI455X (gfx1250, wave32, WMMA).
// f16 activations end-to-end (halves HBM traffic on a memory-bound graph net),
// fused gather + f16-WMMA GEMM + bias + LayerNorm + ReLU per layer,
// async global->LDS staging of gathered tiles, packed-f16 atomics for
// segment sums.
// ---------------------------------------------------------------------------

typedef __attribute__((ext_vector_type(16))) _Float16 v16h;
typedef __attribute__((ext_vector_type(8)))  _Float16 v8h;
typedef __attribute__((ext_vector_type(8)))  float    v8f;
typedef __attribute__((ext_vector_type(4)))  int      v4i;

#define ASTR 400   // LDS A-tile row stride in halves (800B: 16B-aligned, bank-skewed)

// ---- CDNA5 async global->LDS copy (ASYNCcnt path), with safe fallback ------
#if defined(__has_builtin)
#if __has_builtin(__builtin_amdgcn_global_load_async_to_lds_b128)
#define USE_ASYNC_LDS 1
#endif
#endif
#ifndef USE_ASYNC_LDS
#define USE_ASYNC_LDS 0
#endif

typedef __attribute__((address_space(1))) v4i* gv4i_ptr;   // global v4i*
typedef __attribute__((address_space(3))) v4i* lv4i_ptr;   // LDS v4i*

__device__ __forceinline__ void copy16_g2l(const void* g, void* l) {
#if USE_ASYNC_LDS
    __builtin_amdgcn_global_load_async_to_lds_b128((gv4i_ptr)g, (lv4i_ptr)l, 0, 0);
#else
    *(float4*)l = *(const float4*)g;
#endif
}

__device__ __forceinline__ void async_wait0() {
#if USE_ASYNC_LDS
#if __has_builtin(__builtin_amdgcn_s_wait_asynccnt)
    __builtin_amdgcn_s_wait_asynccnt(0);
#else
    asm volatile("s_wait_asynccnt 0x0" ::: "memory");
#endif
#endif
}

// ---------------------------------------------------------------------------
// Pack fp32 weights W[K][128] into per-lane-contiguous f16 fragments matching
// the A fragments' contiguous k order (not AMD's table order): lane l holds col n=l&15,
// element e (0..15) -> k = s*32 + (l>>4)*16 + e.  Zero-padded to KPad.
// ---------------------------------------------------------------------------
__global__ void pack_w_kernel(const float* __restrict__ W, _Float16* __restrict__ Wp,
                              int Kdim, int KPad) {
    const int KSteps = KPad >> 5;
    const long total = (long)KPad * 128;
    long t = (long)blockIdx.x * 256 + threadIdx.x;
    if (t >= total) return;
    int e = (int)(t & 15);
    int l = (int)((t >> 4) & 31);
    long cs = t >> 9;                              // c*KSteps + s
    int s = (int)(cs % KSteps);
    int c = (int)(cs / KSteps);
    int col = c * 16 + (l & 15);
    int k = s * 32 + (l >> 4) * 16 + e;
    Wp[t] = (k < Kdim) ? (_Float16)W[(long)k * 128 + col] : (_Float16)0.f;
}

// ---------------------------------------------------------------------------
// Fused: Y[r,0:128] = ReLU(LN(concat(seg0[i0[r]], seg1[i1[r]], seg2[i2[r]]) @ W + b))
// Segments are f16 (fast async path when width==128) or f32 (bit set in f32mask).
// 16 rows/block, 8 waves, wave w computes output cols [16w,16w+16).
// ---------------------------------------------------------------------------
__global__ __launch_bounds__(256)
void gemm_ln_relu_kernel(const void* __restrict__ X0, const int* __restrict__ idx0, int w0,
                         const void* __restrict__ X1, const int* __restrict__ idx1, int w1,
                         const void* __restrict__ X2, const int* __restrict__ idx2, int w2,
                         int f32mask,
                         const _Float16* __restrict__ Wp,
                         const float* __restrict__ bias,
                         const float* __restrict__ gamma,
                         const float* __restrict__ beta,
                         _Float16* __restrict__ Y, int rows, int KPad) {
    __shared__ _Float16 Atile[16 * ASTR];          // 12.8 KB
    __shared__ float    Crow[16 * 129];            // 8.3 KB
    __shared__ float    mu_s[16], rs_s[16];

    const int tid  = threadIdx.x;
    const int base = blockIdx.x * 16;
    const int KSteps = KPad >> 5;

    const void* Xs[3] = {X0, X1, X2};
    const int*  Is[3] = {idx0, idx1, idx2};
    const int   Wd[3] = {w0, w1, w2};

    // ---- stage gathered/concatenated A tile (16 x KPad, f16) into LDS ----
    bool any_fast = false;
    int segk = 0;
#pragma unroll
    for (int s = 0; s < 3; ++s) {
        const int wsd = Wd[s];
        if (wsd == 0) continue;
        const bool isf32 = (f32mask >> s) & 1;
        if (!isf32 && wsd == 128) {
            // fast path: 16 rows x 16 chunks of 16B == 256 copies == blockDim
            any_fast = true;
            int m = tid >> 4, j = tid & 15;
            long gr = base + m;
            _Float16* ldst = &Atile[m * ASTR + segk + j * 8];
            if (gr < rows) {
                int sr = Is[s] ? Is[s][gr] : (int)gr;
                const _Float16* gsrc = (const _Float16*)Xs[s] + (long)sr * 128 + j * 8;
                copy16_g2l(gsrc, ldst);
            } else {
                unsigned* z = (unsigned*)ldst;
                z[0] = 0u; z[1] = 0u; z[2] = 0u; z[3] = 0u;
            }
        } else {
            for (int t = tid; t < 16 * wsd; t += 256) {
                int m = t / wsd, k = t - m * wsd;
                long gr = base + m;
                float v = 0.f;
                if (gr < rows) {
                    int sr = Is[s] ? Is[s][gr] : (int)gr;
                    v = isf32 ? ((const float*)Xs[s])[(long)sr * wsd + k]
                              : (float)((const _Float16*)Xs[s])[(long)sr * wsd + k];
                }
                Atile[m * ASTR + segk + k] = (_Float16)v;
            }
        }
        segk += wsd;
    }
    if (segk < KPad) {                 // zero-pad K (only the width-3 layer)
        int pad = KPad - segk;
        for (int t = tid; t < 16 * pad; t += 256) {
            int m = t / pad, k = t - m * pad;
            Atile[m * ASTR + segk + k] = (_Float16)0.f;
        }
    }
    if (any_fast) async_wait0();
    __syncthreads();

    const int wv   = tid >> 5;       // wave id = output col tile
    const int lane = tid & 31;
    const int mrow = lane & 15;
    const int hgrp = lane >> 4;

    v8f acc = {0.f, 0.f, 0.f, 0.f, 0.f, 0.f, 0.f, 0.f};
    const _Float16* wbase = Wp + ((long)(wv * KSteps) * 32 + lane) * 16;

    for (int s = 0; s < KSteps; ++s) {
        // A fragment (matches packed B): e -> k = s*32 + hgrp*16 + e
        const _Float16* arow = &Atile[mrow * ASTR + s * 32 + hgrp * 16];
        v16h a;
#pragma unroll
        for (int e = 0; e < 8; ++e) {
            a[e]     = arow[e];
            a[e + 8] = arow[8 + e];
        }
        v16h b = *(const v16h*)(wbase + (long)s * 32 * 16);
        acc = __builtin_amdgcn_wmma_f32_16x16x32_f16(
            false, a, false, b, (short)0, acc, false, false);
    }

    // ---- spill C + bias to LDS for the row-wise LayerNorm ----
#pragma unroll
    for (int r = 0; r < 8; ++r) {
        int m   = r + hgrp * 8;
        int col = wv * 16 + mrow;
        Crow[m * 129 + col] = acc[r] + bias[col];
    }
    __syncthreads();

    if (tid < 16) {
        float s1 = 0.f, s2 = 0.f;
        for (int c = 0; c < 128; ++c) {
            float v = Crow[tid * 129 + c];
            s1 += v; s2 += v * v;
        }
        float mu  = s1 * (1.f / 128.f);
        float var = s2 * (1.f / 128.f) - mu * mu;
        mu_s[tid] = mu;
        rs_s[tid] = rsqrtf(var + 1e-5f);
    }
    __syncthreads();

    {   // normalize + affine + ReLU + one 16B f16 store per thread
        int m  = tid >> 4;
        int c0 = (tid & 15) * 8;
        long gr = base + m;
        if (gr < rows) {
            float mu = mu_s[m], rs = rs_s[m];
            v8h o;
#pragma unroll
            for (int j = 0; j < 8; ++j) {
                int col = c0 + j;
                float v = (Crow[m * 129 + col] - mu) * rs * gamma[col] + beta[col];
                o[j] = (_Float16)fmaxf(v, 0.f);
            }
            *(v8h*)&Y[gr * 128 + c0] = o;
        }
    }
}

// ---------------------------------------------------------------------------
__global__ void fill0_u32_kernel(unsigned* __restrict__ p, long n) {
    long i = (long)blockIdx.x * blockDim.x + threadIdx.x;
    if (i < n) p[i] = 0u;
}

__global__ void f32_to_f16_kernel(const float* __restrict__ s, _Float16* __restrict__ d, long n) {
    long i = (long)blockIdx.x * blockDim.x + threadIdx.x;
    if (i < n) d[i] = (_Float16)s[i];
}

// segment_sum via hardware packed-f16 atomics (global_atomic_pk_add_f16)
__global__ void scatter_add_kernel(const __half2* __restrict__ e2,
                                   const int* __restrict__ idx,
                                   __half2* __restrict__ agg2, long total2) {
    long t = (long)blockIdx.x * 256 + threadIdx.x;
    if (t >= total2) return;
    int edge = (int)(t >> 6);
    int c    = (int)(t & 63);
    unsafeAtomicAdd(&agg2[(long)idx[edge] * 64 + c], e2[t]);
}

// Final 128 -> 1 linear layer: one wave per row, shuffle reduction.
__global__ __launch_bounds__(256)
void final_dot_kernel(const _Float16* __restrict__ X, const float* __restrict__ w,
                      const float* __restrict__ b, float* __restrict__ out, int rows) {
    int wv   = threadIdx.x >> 5;
    int lane = threadIdx.x & 31;
    int row  = blockIdx.x * 8 + wv;
    if (row >= rows) return;
    const _Float16* xr = X + (long)row * 128;
    float s = 0.f;
#pragma unroll
    for (int j = 0; j < 4; ++j) s += (float)xr[lane * 4 + j] * w[lane * 4 + j];
#pragma unroll
    for (int off = 16; off; off >>= 1) s += __shfl_down(s, off, 32);
    if (lane == 0) out[row] = s + b[0];
}

// ---------------------------------------------------------------------------
// d_in layout (setup_inputs dict order; params flatten per-layer as W,b,be,g):
//  [0] x; [1..8] node_enc; [9..16] edge_enc; [17..24] edge_net;
//  [25..32] node_net; [33..44] out_net (3 layers); [n-2] start, [n-1] end.
// ---------------------------------------------------------------------------
extern "C" void kernel_launch(void* const* d_in, const int* in_sizes, int n_in,
                              void* d_out, int out_size, void* d_ws, size_t ws_size,
                              hipStream_t stream) {
    (void)out_size; (void)ws_size;
    const int N = in_sizes[0] / 3;
    const int E = in_sizes[n_in - 2];

    const float* x   = (const float*)d_in[0];
    const int* start = (const int*)d_in[n_in - 2];
    const int* endi  = (const int*)d_in[n_in - 1];
    auto P = [&](int i) { return (const float*)d_in[i]; };

    // ---- workspace carve-out (all activations f16) ----
    char* ws = (char*)d_ws;
    size_t off = 0;
    auto alloc = [&](size_t bytes) -> char* {
        char* p = ws + off;
        off = (off + bytes + 255) & ~(size_t)255;
        return p;
    };
    _Float16* x16  = (_Float16*)alloc((size_t)N * 3 * 2);
    _Float16* h    = (_Float16*)alloc((size_t)N * 128 * 2);
    _Float16* ebuf = (_Float16*)alloc((size_t)E * 128 * 2);
    _Float16* tmp  = (_Float16*)alloc((size_t)E * 128 * 2);
    _Float16* aggE = (_Float16*)alloc((size_t)N * 128 * 2);
    _Float16* aggS = (_Float16*)alloc((size_t)N * 128 * 2);

    const int widx[10]  = {1, 5, 9, 13, 17, 21, 25, 29, 33, 37};
    const int kdims[10] = {3, 128, 256, 128, 384, 128, 384, 128, 384, 128};
    _Float16* pw[10];
    for (int i = 0; i < 10; ++i) {
        int KPad = (kdims[i] + 31) & ~31;
        pw[i] = (_Float16*)alloc((size_t)KPad * 128 * 2);
        long total = (long)KPad * 128;
        pack_w_kernel<<<(int)((total + 255) / 256), 256, 0, stream>>>(
            P(widx[i]), pw[i], kdims[i], KPad);
    }

    f32_to_f16_kernel<<<(int)(((long)N * 3 + 255) / 256), 256, 0, stream>>>(
        x, x16, (long)N * 3);

    auto gemm = [&](const void* X0, const int* i0, int w0,
                    const void* X1, const int* i1, int w1,
                    const void* X2, const int* i2, int w2,
                    int li, _Float16* Y, int rows) {
        int bi = widx[li];
        int KPad = (kdims[li] + 31) & ~31;
        gemm_ln_relu_kernel<<<(rows + 15) / 16, 256, 0, stream>>>(
            X0, i0, w0, X1, i1, w1, X2, i2, w2, /*f32mask=*/0,
            pw[li], P(bi + 1) /*b*/, P(bi + 3) /*g*/, P(bi + 2) /*be*/,
            Y, rows, KPad);
    };

    // ---- node encoder: x16[N,3] -> h[N,128] ----
    gemm(x16, nullptr, 3,   nullptr, nullptr, 0, nullptr, nullptr, 0, 0, tmp, N);
    gemm(tmp, nullptr, 128, nullptr, nullptr, 0, nullptr, nullptr, 0, 1, h,   N);

    // ---- edge encoder: [h[start], h[end]] -> e[E,128] ----
    gemm(h,   start,   128, h, endi, 128, nullptr, nullptr, 0, 2, tmp,  E);
    gemm(tmp, nullptr, 128, nullptr, nullptr, 0, nullptr, nullptr, 0, 3, ebuf, E);

    const long aggw  = (long)N * 64;   // half2 words
    const long etot2 = (long)E * 64;   // half2 elements
    for (int it = 0; it < 3; ++it) {
        fill0_u32_kernel<<<(int)((aggw + 255) / 256), 256, 0, stream>>>((unsigned*)aggE, aggw);
        fill0_u32_kernel<<<(int)((aggw + 255) / 256), 256, 0, stream>>>((unsigned*)aggS, aggw);
        scatter_add_kernel<<<(int)((etot2 + 255) / 256), 256, 0, stream>>>(
            (const __half2*)ebuf, endi,  (__half2*)aggE, etot2);
        scatter_add_kernel<<<(int)((etot2 + 255) / 256), 256, 0, stream>>>(
            (const __half2*)ebuf, start, (__half2*)aggS, etot2);
        // node_net: [h, aggE, aggS] -> h
        gemm(h,   nullptr, 128, aggE, nullptr, 128, aggS, nullptr, 128, 6, tmp, N);
        gemm(tmp, nullptr, 128, nullptr, nullptr, 0, nullptr, nullptr, 0, 7, h,  N);
        // edge_net: [h[start], h[end], e] -> e
        gemm(h,   start,   128, h, endi, 128, ebuf, nullptr, 128, 4, tmp,  E);
        gemm(tmp, nullptr, 128, nullptr, nullptr, 0, nullptr, nullptr, 0, 5, ebuf, E);
    }

    // ---- output classifier ----
    gemm(h,   start,   128, h, endi, 128, ebuf, nullptr, 128, 8, tmp, E);
    gemm(tmp, nullptr, 128, nullptr, nullptr, 0, nullptr, nullptr, 0, 9, tmp, E); // in-place safe
    final_dot_kernel<<<(E + 7) / 8, 256, 0, stream>>>(tmp, P(41), P(42), (float*)d_out, E);
}
